// JambaMambaDecoderLayer_60782377173434
// MI455X (gfx1250) — matrix operand order repaired and checked by reading, not compiled, not run on hardware
//
#include <hip/hip_runtime.h>
#include <cstddef>

// ---------- types ----------
typedef __attribute__((ext_vector_type(8)))  float    v8f;
typedef __attribute__((ext_vector_type(16))) __bf16   v16bf;
typedef __attribute__((ext_vector_type(2)))  __bf16   v2bf;
typedef __attribute__((ext_vector_type(4)))  unsigned u32x4;
typedef __attribute__((ext_vector_type(2)))  unsigned u32x2;
typedef __attribute__((ext_vector_type(4)))  float    f32x4;
typedef __attribute__((ext_vector_type(2)))  float    f32x2;

struct Frag2 { u32x4 lo, hi; };   // 32 bytes == v16bf

static __device__ __forceinline__ unsigned bf16pair(float a, float b) {
    // fptrunc <2 x float> -> <2 x bfloat>  ==>  single v_cvt_pk_bf16_f32
    f32x2 in = {a, b};
    v2bf  r  = __builtin_convertvector(in, v2bf);
    return __builtin_bit_cast(unsigned, r);
}

static __device__ __forceinline__ float sigmoidf_(float x) {
    return 1.0f / (1.0f + __expf(-x));
}

// ---------- sizes ----------
#define S_   1024
#define H_   2048
#define I_   4096
#define N_   16
#define KC_  4
#define DTR_ 128
#define F_   5632
#define P_   160   // DTR + 2N

// =====================================================================
// GEMM: C[m,n] = sum_k A[m,k] * W[n,k]   (A: MxK fp32, W: NxK fp32, C: MxN fp32)
// M = gridDim.y*128 (exact), K/32 even.
// GUARDN: N ragged (branch-free clamp+select); else N multiple of 128.
// bf16 WMMA 16x16x32, 128x128 block tile, 8 waves, each wave 64x32.
// Double-buffered LDS, ONE barrier per K-step; next tile's global loads
// and its cvt+ds_store overlap the WMMA block.
// =====================================================================
template <bool GUARDN>
__global__ __launch_bounds__(256) void k_gemm_bf16(
    const float* __restrict__ A, const float* __restrict__ W,
    float* __restrict__ C, int Ncols, int Kdim)
{
    __shared__ __align__(16) unsigned sA[2][128 * 16]; // packed bf16 pairs
    __shared__ __align__(16) unsigned sW[2][128 * 16];

    const int tid  = threadIdx.x;
    const int m0   = blockIdx.y * 128;
    const int n0   = blockIdx.x * 128;
    const int wave = tid >> 5;
    const int lane = tid & 31;
    const int lr   = lane & 15;   // row within 16x16 tile
    const int lh   = lane >> 4;   // lane half
    const int wm   = (wave & 1) * 64;
    const int wn   = (wave >> 1) * 32;

    // per-thread global load coordinates (4 x f32x4 per matrix per K-step)
    int ldr[4], ldc4[4];
    #pragma unroll
    for (int q = 0; q < 4; ++q) {
        int idx = tid + q * 256;          // 0..1023
        ldr[q]  = idx >> 3;               // 0..127
        ldc4[q] = (idx & 7) * 4;          // k offset within tile
    }

    v8f acc[4][2];
    #pragma unroll
    for (int im = 0; im < 4; ++im)
        #pragma unroll
        for (int in2 = 0; in2 < 2; ++in2)
            acc[im][in2] = v8f{0.f,0.f,0.f,0.f,0.f,0.f,0.f,0.f};

    f32x4 av[4], wv[4];

    auto load_tile = [&](int ktb) {
        #pragma unroll
        for (int q = 0; q < 4; ++q) {
            av[q] = *reinterpret_cast<const f32x4*>(
                        A + (size_t)(m0 + ldr[q]) * Kdim + ktb + ldc4[q]);
            if (GUARDN) {
                int wrow = n0 + ldr[q];
                int wr   = (wrow < Ncols) ? wrow : (Ncols - 1);
                f32x4 t  = *reinterpret_cast<const f32x4*>(
                               W + (size_t)wr * Kdim + ktb + ldc4[q]);
                wv[q] = (wrow < Ncols) ? t : f32x4{0.f, 0.f, 0.f, 0.f};
            } else {
                wv[q] = *reinterpret_cast<const f32x4*>(
                            W + (size_t)(n0 + ldr[q]) * Kdim + ktb + ldc4[q]);
            }
        }
    };

    auto store_tile = [&](int p) {
        #pragma unroll
        for (int q = 0; q < 4; ++q) {
            int r  = ldr[q];
            int c2 = ldc4[q] >> 1;        // packed-pair column (even)
            *reinterpret_cast<u32x2*>(&sA[p][r * 16 + c2]) =
                u32x2{bf16pair(av[q].x, av[q].y), bf16pair(av[q].z, av[q].w)};
            *reinterpret_cast<u32x2*>(&sW[p][r * 16 + c2]) =
                u32x2{bf16pair(wv[q].x, wv[q].y), bf16pair(wv[q].z, wv[q].w)};
        }
    };

    auto step = [&](int p, int kt) {
        __syncthreads();  // publishes buf[p]; all reads of buf[p^1] are done
        bool more = (kt + 32) < Kdim;
        if (more) {
            load_tile(kt + 32);
            if (kt + 64 < Kdim) {   // prime L2 two tiles ahead
                __builtin_prefetch(A + (size_t)(m0 + ldr[0]) * Kdim + kt + 64, 0, 1);
                __builtin_prefetch(W + (size_t)(n0 + (GUARDN ? 0 : ldr[0])) * Kdim
                                     + kt + 64, 0, 1);
            }
        }
        // ---- fragment gathers + 8 WMMAs from buffer p ----
        v16bf af[4], bf[2];
        #pragma unroll
        for (int im = 0; im < 4; ++im) {
            int r = wm + im * 16 + lr;
            Frag2 f;
            f.lo = *reinterpret_cast<const u32x4*>(&sA[p][r * 16 + lh * 4]);
            f.hi = *reinterpret_cast<const u32x4*>(&sA[p][r * 16 + 8 + lh * 4]);
            af[im] = __builtin_bit_cast(v16bf, f);
        }
        #pragma unroll
        for (int in2 = 0; in2 < 2; ++in2) {
            int r = wn + in2 * 16 + lr;
            Frag2 f;
            f.lo = *reinterpret_cast<const u32x4*>(&sW[p][r * 16 + lh * 4]);
            f.hi = *reinterpret_cast<const u32x4*>(&sW[p][r * 16 + 8 + lh * 4]);
            bf[in2] = __builtin_bit_cast(v16bf, f);
        }
        #pragma unroll
        for (int im = 0; im < 4; ++im)
            #pragma unroll
            for (int in2 = 0; in2 < 2; ++in2)
                acc[im][in2] = __builtin_amdgcn_wmma_f32_16x16x32_bf16(
                    false, af[im], false, bf[in2], (short)0, acc[im][in2],
                    false, false);
        // ---- convert + stage next tile into the other buffer ----
        if (more) store_tile(p ^ 1);
    };

    // prologue: tile 0 -> buffer 0 (published by first step's barrier)
    load_tile(0);
    store_tile(0);

    for (int kt = 0; kt < Kdim; kt += 64) {   // K/32 is even for all our GEMMs
        step(0, kt);
        step(1, kt + 32);
    }

    #pragma unroll
    for (int im = 0; im < 4; ++im) {
        #pragma unroll
        for (int in2 = 0; in2 < 2; ++in2) {
            int n = n0 + wn + in2 * 16 + lr;
            if (!GUARDN || n < Ncols) {
                #pragma unroll
                for (int v = 0; v < 8; ++v) {
                    int m = m0 + wm + im * 16 + v + lh * 8;
                    C[(size_t)m * Ncols + n] = acc[im][in2][v];
                }
            }
        }
    }
}

// =====================================================================
// rmsnorm: one block per row
// =====================================================================
__global__ __launch_bounds__(256) void k_rmsnorm(
    const float* __restrict__ x, const float* __restrict__ w,
    float* __restrict__ out, int ncols)
{
    int row = blockIdx.x, t = threadIdx.x;
    const float* xr = x + (size_t)row * ncols;
    float ss = 0.f;
    for (int c = t; c < ncols; c += 256) { float v = xr[c]; ss += v * v; }
    __shared__ float red[256];
    red[t] = ss; __syncthreads();
    for (int s = 128; s > 0; s >>= 1) {
        if (t < s) red[t] += red[t + s];
        __syncthreads();
    }
    float scale = rsqrtf(red[0] / (float)ncols + 1e-6f);
    float* orow = out + (size_t)row * ncols;
    for (int c = t; c < ncols; c += 256) orow[c] = xr[c] * scale * w[c];
}

// =====================================================================
// residual add + rmsnorm (H=2048 fixed): resid = mix + hid; h2 = rmsnorm(resid)
// =====================================================================
__global__ __launch_bounds__(256) void k_res_rms(
    const float* __restrict__ mix, const float* __restrict__ hid,
    const float* __restrict__ w, float* __restrict__ resid,
    float* __restrict__ h2)
{
    int row = blockIdx.x, t = threadIdx.x;
    size_t base = (size_t)row * H_;
    float r8[8]; float ss = 0.f;
    #pragma unroll
    for (int q = 0; q < 8; ++q) {
        int c = t + q * 256;
        float v = mix[base + c] + hid[base + c];
        r8[q] = v; resid[base + c] = v; ss += v * v;
    }
    __shared__ float red[256];
    red[t] = ss; __syncthreads();
    for (int s = 128; s > 0; s >>= 1) {
        if (t < s) red[t] += red[t + s];
        __syncthreads();
    }
    float scale = rsqrtf(red[0] / (float)H_ + 1e-6f);
    #pragma unroll
    for (int q = 0; q < 8; ++q) {
        int c = t + q * 256;
        h2[base + c] = r8[q] * scale * w[c];
    }
}

// =====================================================================
// causal depthwise conv (K=4) + silu.  x = xz[:, :I]
// =====================================================================
__global__ __launch_bounds__(256) void k_conv_silu(
    const float* __restrict__ xz, const float* __restrict__ cw,
    const float* __restrict__ cb, float* __restrict__ xc)
{
    int idx = blockIdx.x * 256 + threadIdx.x;  // s*I + i
    int s = idx >> 12;          // / I_
    int i = idx & (I_ - 1);
    f32x4 w4 = *reinterpret_cast<const f32x4*>(cw + (size_t)i * KC_);
    float acc = cb[i];
    #pragma unroll
    for (int k = 0; k < KC_; ++k) {
        int sp = s - (KC_ - 1) + k;
        if (sp >= 0) acc += w4[k] * xz[(size_t)sp * (2 * I_) + i];
    }
    xc[idx] = acc * sigmoidf_(acc);
}

// =====================================================================
// rmsnorm of dt(128) / B(16) / C(16) segments of ssm_p rows
// =====================================================================
__global__ __launch_bounds__(256) void k_norm_dtbc(
    const float* __restrict__ ssm, const float* __restrict__ wdt,
    const float* __restrict__ wb, const float* __restrict__ wc,
    float* __restrict__ dtN, float* __restrict__ Bn, float* __restrict__ Cn)
{
    int s = blockIdx.x, t = threadIdx.x;
    __shared__ float sh[P_];
    __shared__ float scl[3];
    if (t < P_) sh[t] = ssm[(size_t)s * P_ + t];
    __syncthreads();
    if (t == 0) { float a = 0; for (int j = 0;   j < 128; ++j) a += sh[j]*sh[j]; scl[0] = rsqrtf(a / 128.f + 1e-6f); }
    if (t == 1) { float a = 0; for (int j = 128; j < 144; ++j) a += sh[j]*sh[j]; scl[1] = rsqrtf(a / 16.f  + 1e-6f); }
    if (t == 2) { float a = 0; for (int j = 144; j < 160; ++j) a += sh[j]*sh[j]; scl[2] = rsqrtf(a / 16.f  + 1e-6f); }
    __syncthreads();
    if (t < 128)      dtN[(size_t)s * DTR_ + t]      = sh[t] * scl[0] * wdt[t];
    else if (t < 144) Bn [(size_t)s * N_ + (t - 128)] = sh[t] * scl[1] * wb[t - 128];
    else if (t < 160) Cn [(size_t)s * N_ + (t - 144)] = sh[t] * scl[2] * wc[t - 144];
}

// =====================================================================
// softplus(x + bias)  in place on delta (S x I)
// =====================================================================
__global__ __launch_bounds__(256) void k_softplus(
    float* __restrict__ delta, const float* __restrict__ bias)
{
    int idx = blockIdx.x * 256 + threadIdx.x;
    int i = idx & (I_ - 1);
    float v = delta[idx] + bias[i];
    delta[idx] = (v > 20.f) ? v : log1pf(__expf(v));
}

// =====================================================================
// selective scan: one thread per (channel i, state n); shfl-reduce the
// 16-wide state*C dot within each 16-lane group (wave32).
// y = (scan_out + xc*D) * silu(z)
// =====================================================================
__global__ __launch_bounds__(256) void k_scan(
    const float* __restrict__ delta, const float* __restrict__ xc,
    const float* __restrict__ xzfull, const float* __restrict__ Bn,
    const float* __restrict__ Cn, const float* __restrict__ Aw,
    const float* __restrict__ Dw, float* __restrict__ y)
{
    int g = blockIdx.x * 256 + threadIdx.x;  // 0 .. I*N-1
    int i = g >> 4;
    int n = g & 15;
    float a  = Aw[(size_t)i * N_ + n];
    float di = Dw[i];
    float st = 0.f;
    for (int s = 0; s < S_; ++s) {
        float dt = delta[(size_t)s * I_ + i];
        float u  = xc[(size_t)s * I_ + i];
        float b  = Bn[(size_t)s * N_ + n];
        float c  = Cn[(size_t)s * N_ + n];
        st = st * __expf(dt * a) + (dt * u) * b;
        float p = st * c;
        p += __shfl_xor(p, 1);
        p += __shfl_xor(p, 2);
        p += __shfl_xor(p, 4);
        p += __shfl_xor(p, 8);
        if (n == 0) {
            float zv = xzfull[(size_t)s * (2 * I_) + I_ + i];
            y[(size_t)s * I_ + i] = (p + u * di) * (zv * sigmoidf_(zv));
        }
    }
}

// =====================================================================
// GLU: hf = silu(gu[:, :F]) * gu[:, F:]
// =====================================================================
__global__ __launch_bounds__(256) void k_glu(
    const float* __restrict__ gu, float* __restrict__ hf)
{
    int idx = blockIdx.x * 256 + threadIdx.x;   // s*F + f
    int s = idx / F_;
    int f = idx - s * F_;
    float g = gu[(size_t)s * (2 * F_) + f];
    float u = gu[(size_t)s * (2 * F_) + F_ + f];
    hf[idx] = g * sigmoidf_(g) * u;
}

// =====================================================================
extern "C" void kernel_launch(void* const* d_in, const int* in_sizes, int n_in,
                              void* d_out, int out_size, void* d_ws, size_t ws_size,
                              hipStream_t stream)
{
    (void)in_sizes; (void)n_in; (void)out_size; (void)ws_size;

    const float* hidden    = (const float*)d_in[0];
    const float* ln_in_w   = (const float*)d_in[1];
    const float* in_proj_w = (const float*)d_in[2];   // [2I, H]
    const float* conv_w    = (const float*)d_in[3];   // [I, K]
    const float* conv_b    = (const float*)d_in[4];
    const float* x_proj_w  = (const float*)d_in[5];   // [P, I]
    const float* dt_ln_w   = (const float*)d_in[6];
    const float* b_ln_w    = (const float*)d_in[7];
    const float* c_ln_w    = (const float*)d_in[8];
    const float* dt_proj_w = (const float*)d_in[9];   // [I, DTR]
    const float* dt_proj_b = (const float*)d_in[10];
    const float* Aw        = (const float*)d_in[11];  // [I, N]
    const float* Dw        = (const float*)d_in[12];
    const float* out_proj_w= (const float*)d_in[13];  // [H, I]
    const float* ln_ff_w   = (const float*)d_in[14];
    const float* gate_up_w = (const float*)d_in[15];  // [2F, H]
    const float* down_w    = (const float*)d_in[16];  // [H, F]

    float* ws    = (float*)d_ws;
    float* hN    = ws;                                   // S*H
    float* xz    = hN    + (size_t)S_ * H_;              // S*2I
    float* xc    = xz    + (size_t)S_ * 2 * I_;          // S*I
    float* ssm   = xc    + (size_t)S_ * I_;              // S*P
    float* dtN   = ssm   + (size_t)S_ * P_;              // S*DTR
    float* Bn    = dtN   + (size_t)S_ * DTR_;            // S*N
    float* Cn    = Bn    + (size_t)S_ * N_;              // S*N
    float* delta = Cn    + (size_t)S_ * N_;              // S*I
    float* yAct  = delta + (size_t)S_ * I_;              // S*I
    float* mix   = yAct  + (size_t)S_ * I_;              // S*H
    float* h2    = mix   + (size_t)S_ * H_;              // S*H
    float* gu    = xz;     // reuse xz+xc region (dead by then): S*2F
    float* hf    = delta;  // reuse delta+yAct region:          S*F

    float* out_h   = (float*)d_out;
    float* out_res = out_h + (size_t)S_ * H_;

    // 1) rmsnorm input
    k_rmsnorm<<<dim3(S_), dim3(256), 0, stream>>>(hidden, ln_in_w, hN, H_);
    // 2) xz = hN @ in_proj^T   (M=1024, N=8192, K=2048)
    k_gemm_bf16<false><<<dim3(2 * I_ / 128, S_ / 128), dim3(256), 0, stream>>>(
        hN, in_proj_w, xz, 2 * I_, H_);
    // 3) causal conv + silu
    k_conv_silu<<<dim3(S_ * I_ / 256), dim3(256), 0, stream>>>(xz, conv_w, conv_b, xc);
    // 4) ssm_p = xc @ x_proj^T   (N=160, K=4096)  -- ragged N
    k_gemm_bf16<true><<<dim3((P_ + 127) / 128, S_ / 128), dim3(256), 0, stream>>>(
        xc, x_proj_w, ssm, P_, I_);
    // 5) per-segment rmsnorms
    k_norm_dtbc<<<dim3(S_), dim3(256), 0, stream>>>(ssm, dt_ln_w, b_ln_w, c_ln_w,
                                                    dtN, Bn, Cn);
    // 6) delta = dtN @ dt_proj^T   (N=4096, K=128)
    k_gemm_bf16<false><<<dim3(I_ / 128, S_ / 128), dim3(256), 0, stream>>>(
        dtN, dt_proj_w, delta, I_, DTR_);
    // 7) softplus(+bias) in place
    k_softplus<<<dim3(S_ * I_ / 256), dim3(256), 0, stream>>>(delta, dt_proj_b);
    // 8) selective scan -> y*(silu(z)) activation
    k_scan<<<dim3(I_ * N_ / 256), dim3(256), 0, stream>>>(delta, xc, xz, Bn, Cn,
                                                          Aw, Dw, yAct);
    // 9) mix = yAct @ out_proj^T   (N=2048, K=4096)
    k_gemm_bf16<false><<<dim3(H_ / 128, S_ / 128), dim3(256), 0, stream>>>(
        yAct, out_proj_w, mix, H_, I_);
    // 10) residual add + rmsnorm
    k_res_rms<<<dim3(S_), dim3(256), 0, stream>>>(mix, hidden, ln_ff_w, out_res, h2);
    // 11) gu = h2 @ gate_up^T   (N=11264, K=2048)
    k_gemm_bf16<false><<<dim3(2 * F_ / 128, S_ / 128), dim3(256), 0, stream>>>(
        h2, gate_up_w, gu, 2 * F_, H_);
    // 12) GLU
    k_glu<<<dim3(S_ * F_ / 256), dim3(256), 0, stream>>>(gu, hf);
    // 13) out = hf @ down^T   (N=2048, K=5632)
    k_gemm_bf16<false><<<dim3(H_ / 128, S_ / 128), dim3(256), 0, stream>>>(
        hf, down_w, out_h, H_, F_);
}
